// StandardAttention_14894946582817
// MI455X (gfx1250) — matrix operand order repaired and checked by reading, not compiled, not run on hardware
//
#include <hip/hip_runtime.h>

typedef __attribute__((ext_vector_type(16))) __bf16 v16bf;
typedef __attribute__((ext_vector_type(8)))  float  v8f;

union Pk2 { unsigned int u; __bf16 h[2]; };

__device__ inline unsigned int pack2(float a, float b) {
    Pk2 p; p.h[0] = (__bf16)a; p.h[1] = (__bf16)b; return p.u;
}

__device__ inline unsigned lds_addr_of(const void* p) {
    // shared-aperture generic pointers keep the LDS byte offset in addr[31:0]
    return (unsigned)(unsigned long long)p;
}

// async global->LDS copy of 16 bytes per lane (ASYNCcnt-tracked, gfx1250)
__device__ inline void async_copy_b128(unsigned lds_byte_addr, unsigned global_byte_off,
                                       const void* base) {
    asm volatile("global_load_async_to_lds_b128 %0, %1, %2"
                 :: "v"(lds_byte_addr), "v"(global_byte_off),
                    "s"((unsigned long long)base)
                 : "memory");
}
__device__ inline void wait_async0() {
    asm volatile("s_wait_asynccnt 0x0" ::: "memory");
}

// A-fragment (16xK=32 bf16), ISA layout: lane m = lane&15, lane-half selects K sub-block.
__device__ inline v16bf load_afrag(const __bf16* base, int stride, int lane) {
    const int m = lane & 15, hi = lane >> 4;
    v16bf f;
#pragma unroll
    for (int i = 0; i < 8; ++i) {
        const int kb = 2 * (i & 3) + ((i >= 4) ? 16 : 0) + hi * 8;
        Pk2 p; p.u = *(const unsigned int*)(base + m * stride + kb);
        f[2 * i] = p.h[0]; f[2 * i + 1] = p.h[1];
    }
    return f;
}

// B-fragment (K=32 x 16 bf16): lane n = lane&15, lane-half selects K sub-block (same order as A).
// element (k, n) read from base[n*stride + k]   (base holds B^T row-major)
__device__ inline v16bf load_bfrag(const __bf16* base, int stride, int lane) {
    const int n = lane & 15, hi = lane >> 4;
    v16bf f;
#pragma unroll
    for (int i = 0; i < 8; ++i) {
        const int kb = 2 * (i & 3) + ((i >= 4) ? 16 : 0) + hi * 8;
        Pk2 p; p.u = *(const unsigned int*)(base + n * stride + kb);
        f[2 * i] = p.h[0]; f[2 * i + 1] = p.h[1];
    }
    return f;
}

// ---------------------------------------------------------------------------
// fp32 -> bf16 elementwise (vectorized), run once per tensor
// ---------------------------------------------------------------------------
__global__ __launch_bounds__(256) void cvt_f32_bf16(
    const float4* __restrict__ in, uint2* __restrict__ out, int n4)
{
    const int i = blockIdx.x * 256 + threadIdx.x;
    if (i < n4) {
        const float4 v = in[i];
        uint2 o; o.x = pack2(v.x, v.y); o.y = pack2(v.z, v.w);
        out[i] = o;
    }
}

// ---------------------------------------------------------------------------
// Y[M,N] = X[M,K] @ W[N,K]^T + bias[N]      (bf16 in, bf16 or fp32 out)
// 64x64 tile / 128 threads (4 waves); K-step 64; async-to-LDS staging.
// ---------------------------------------------------------------------------
template <int OUT_BF16>
__global__ __launch_bounds__(128) void gemm_bias(
    const __bf16* __restrict__ X, const __bf16* __restrict__ W,
    const float* __restrict__ bias, void* __restrict__ out,
    int M, int N, int K)
{
    __shared__ __bf16 As[64 * 72];   // [m][k0..63] pad 64->72 halves (144B rows)
    __shared__ __bf16 Ws[64 * 72];   // [n][k0..63]
    const int tid = threadIdx.x, lane = tid & 31, wv = tid >> 5;
    const int m0 = blockIdx.x * 64, n0 = blockIdx.y * 64;

    v8f acc[4] = {v8f{}, v8f{}, v8f{}, v8f{}};

    for (int kt = 0; kt < K; kt += 64) {
        // stage 64x64 bf16 tiles of A and W via async copies (2 chunks each/thread)
#pragma unroll
        for (int cc = 0; cc < 2; ++cc) {
            const int c = tid + cc * 128;        // 256 chunks of 16 elems
            const int row = c >> 2, ch = (c & 3) * 16;
            const unsigned aoff = (unsigned)(((size_t)(m0 + row) * K + kt + ch) * 2);
            const unsigned woff = (unsigned)(((size_t)(n0 + row) * K + kt + ch) * 2);
            async_copy_b128(lds_addr_of(&As[row * 72 + ch]), aoff, X);
            async_copy_b128(lds_addr_of(&Ws[row * 72 + ch]), woff, W);
            if (kt + 64 < K) {
                __builtin_prefetch(X + (size_t)(m0 + row) * K + (kt + 64) + ch, 0, 0);
                __builtin_prefetch(W + (size_t)(n0 + row) * K + (kt + 64) + ch, 0, 0);
            }
        }
        wait_async0();
        __syncthreads();

#pragma unroll
        for (int kk = 0; kk < 64; kk += 32) {
            const v16bf bf = load_bfrag(&Ws[(wv * 16) * 72 + kk], 72, lane);
#pragma unroll
            for (int mt = 0; mt < 4; ++mt) {
                const v16bf af = load_afrag(&As[(mt * 16) * 72 + kk], 72, lane);
                acc[mt] = __builtin_amdgcn_wmma_f32_16x16x32_bf16(
                    false, af, false, bf, (short)0, acc[mt], false, false);
            }
        }
        __syncthreads();
    }

    const int n = n0 + wv * 16 + (lane & 15);
    const int hi = lane >> 4;
    const float bv = bias[n];
#pragma unroll
    for (int mt = 0; mt < 4; ++mt)
#pragma unroll
        for (int r = 0; r < 8; ++r) {
            const int m = m0 + mt * 16 + r + 8 * hi;
            const float v = acc[mt][r] + bv;
            if (OUT_BF16) ((__bf16*)out)[(size_t)m * N + n] = (__bf16)v;
            else          ((float*)out)[(size_t)m * N + n] = v;
        }
}

// ---------------------------------------------------------------------------
// Causal flash attention over bf16 q/k/v laid out [B*T, C] (head = 64-col slice).
// Block: 64 query rows (4 waves x 16); 32-wide key blocks, online softmax.
// K tile staged via async-to-LDS; V transposed through VGPRs; y written bf16.
// ---------------------------------------------------------------------------
__global__ __launch_bounds__(128) void flash_attn(
    const __bf16* __restrict__ Q, const __bf16* __restrict__ Kb,
    const __bf16* __restrict__ Vb, __bf16* __restrict__ Y,
    int T, int C)
{
    constexpr int DH = 64;
    __shared__ __bf16 Ks[32 * 72];      // [kc][d]  pad 64->72 (144B rows, 16B aligned)
    __shared__ __bf16 Vt[64 * 36];      // [d][kc]  transposed V
    __shared__ __bf16 Ps[4][16 * 36];   // per-wave P tile [m][kc]

    const int tid = threadIdx.x, lane = tid & 31, wv = tid >> 5;
    const int qBase = blockIdx.x * 64;
    const int b = blockIdx.y >> 4, h = blockIdx.y & 15;
    const int qrow0 = qBase + wv * 16;
    const float scale = 0.125f;  // 1/sqrt(64)
    const size_t headOff = ((size_t)b * T) * C + (size_t)h * DH;

    // Q fragments (2 K-chunks of 32 over DH=64), loaded once from global bf16
    v16bf aq[2];
    {
        const __bf16* qb = Q + headOff + (size_t)qrow0 * C;
        const int m = lane & 15, hi = lane >> 4;
#pragma unroll
        for (int j = 0; j < 2; ++j) {
            v16bf f;
#pragma unroll
            for (int i = 0; i < 8; ++i) {
                const int kb = 2 * (i & 3) + ((i >= 4) ? 16 : 0) + hi * 8 + 32 * j;
                Pk2 p; p.u = *(const unsigned int*)(qb + (size_t)m * C + kb);
                f[2 * i] = p.h[0]; f[2 * i + 1] = p.h[1];
            }
            aq[j] = f;
        }
    }

    v8f o[4] = {v8f{}, v8f{}, v8f{}, v8f{}};
    float mi[8], li[8];
#pragma unroll
    for (int r = 0; r < 8; ++r) { mi[r] = -3.0e38f; li[r] = 0.0f; }

    const int kEnd = qBase + 64;  // causal: later keys fully masked
    for (int kb = 0; kb < kEnd; kb += 32) {
        {   // stage K (async-to-LDS) and V (VGPR transpose): 4 threads/row
            const int kc = tid >> 2, ch = (tid & 3) * 16;
            const unsigned koff =
                (unsigned)((headOff + (size_t)(kb + kc) * C + ch) * 2);
            async_copy_b128(lds_addr_of(&Ks[kc * 72 + ch]), koff, Kb);

            const __bf16* vg = Vb + headOff + (size_t)(kb + kc) * C + ch;
            __bf16 vtmp[16];
            *(uint4*)&vtmp[0] = *(const uint4*)vg;
            *(uint4*)&vtmp[8] = *(const uint4*)(vg + 8);
#pragma unroll
            for (int j = 0; j < 16; ++j) Vt[(ch + j) * 36 + kc] = vtmp[j];
            wait_async0();
        }
        __syncthreads();

        // S = Q K^T : two 16x16 key tiles, K-dim = DH in two chunks of 32
        v8f s0 = v8f{}, s1 = v8f{};
#pragma unroll
        for (int j = 0; j < 2; ++j) {
            const v16bf bk0 = load_bfrag(&Ks[0 * 16 * 72 + 32 * j], 72, lane);
            const v16bf bk1 = load_bfrag(&Ks[1 * 16 * 72 + 32 * j], 72, lane);
            s0 = __builtin_amdgcn_wmma_f32_16x16x32_bf16(false, aq[j], false, bk0,
                                                         (short)0, s0, false, false);
            s1 = __builtin_amdgcn_wmma_f32_16x16x32_bf16(false, aq[j], false, bk1,
                                                         (short)0, s1, false, false);
        }

        // mask + online softmax (row stats per lane-half; 8 rows per lane)
        const int hi = lane >> 4;
        const int col0 = kb + (lane & 15), col1 = col0 + 16;
        float fac[8], pr0[8], pr1[8];
#pragma unroll
        for (int r = 0; r < 8; ++r) {
            const int qr = qrow0 + r + 8 * hi;
            float a = s0[r] * scale; if (col0 > qr) a = -3.0e38f;
            float c = s1[r] * scale; if (col1 > qr) c = -3.0e38f;
            float mx = fmaxf(a, c);
#pragma unroll
            for (int off = 1; off < 16; off <<= 1) mx = fmaxf(mx, __shfl_xor(mx, off));
            const float mn = fmaxf(mi[r], mx);
            const float f = __expf(mi[r] - mn);
            const float p0 = (col0 > qr) ? 0.0f : __expf(a - mn);
            const float p1 = (col1 > qr) ? 0.0f : __expf(c - mn);
            float sum = p0 + p1;
#pragma unroll
            for (int off = 1; off < 16; off <<= 1) sum += __shfl_xor(sum, off);
            mi[r] = mn; li[r] = li[r] * f + sum;
            fac[r] = f; pr0[r] = p0; pr1[r] = p1;
        }
#pragma unroll
        for (int nt = 0; nt < 4; ++nt)
#pragma unroll
            for (int r = 0; r < 8; ++r) o[nt][r] *= fac[r];

        // P -> LDS (C-layout -> row-major), then re-read as A-fragment
#pragma unroll
        for (int r = 0; r < 8; ++r) {
            const int row = r + 8 * hi;
            Ps[wv][row * 36 + (lane & 15)]      = (__bf16)pr0[r];
            Ps[wv][row * 36 + 16 + (lane & 15)] = (__bf16)pr1[r];
        }
        __syncthreads();

        const v16bf ap = load_afrag(&Ps[wv][0], 36, lane);
#pragma unroll
        for (int nt = 0; nt < 4; ++nt) {
            const v16bf bv = load_bfrag(&Vt[(nt * 16) * 36], 36, lane);
            o[nt] = __builtin_amdgcn_wmma_f32_16x16x32_bf16(false, ap, false, bv,
                                                            (short)0, o[nt], false, false);
        }
        __syncthreads();
    }

    // normalize and write y (bf16, [B*T, C])
    const int hi = lane >> 4;
#pragma unroll
    for (int nt = 0; nt < 4; ++nt)
#pragma unroll
        for (int r = 0; r < 8; ++r) {
            const int qr = qrow0 + r + 8 * hi;
            Y[headOff + (size_t)qr * C + nt * 16 + (lane & 15)] =
                (__bf16)(o[nt][r] / li[r]);
        }
}

// ---------------------------------------------------------------------------
extern "C" void kernel_launch(void* const* d_in, const int* in_sizes, int n_in,
                              void* d_out, int out_size, void* d_ws, size_t ws_size,
                              hipStream_t stream) {
    (void)in_sizes; (void)n_in; (void)out_size; (void)ws_size;
    const float* x  = (const float*)d_in[0];
    const float* Wq = (const float*)d_in[1];
    const float* bq = (const float*)d_in[2];
    const float* Wk = (const float*)d_in[3];
    const float* bk = (const float*)d_in[4];
    const float* Wv = (const float*)d_in[5];
    const float* bv = (const float*)d_in[6];
    const float* Wo = (const float*)d_in[7];
    const float* bo = (const float*)d_in[8];

    constexpr int B = 4, T = 2048, C = 1024, H = 16;
    constexpr int M = B * T;                 // 8192
    constexpr size_t NX = (size_t)M * C;     // 8388608
    constexpr size_t NW = (size_t)C * C;     // 1048576

    __bf16* xb  = (__bf16*)d_ws;
    __bf16* wqb = xb  + NX;
    __bf16* wkb = wqb + NW;
    __bf16* wvb = wkb + NW;
    __bf16* wob = wvb + NW;
    __bf16* qb  = wob + NW;
    __bf16* kbf = qb  + NX;
    __bf16* vbf = kbf + NX;
    __bf16* yb  = vbf + NX;

    // one-time fp32 -> bf16 conversions
    cvt_f32_bf16<<<(int)(NX / 4 + 255) / 256, 256, 0, stream>>>((const float4*)x,  (uint2*)xb,  (int)(NX / 4));
    cvt_f32_bf16<<<(int)(NW / 4 + 255) / 256, 256, 0, stream>>>((const float4*)Wq, (uint2*)wqb, (int)(NW / 4));
    cvt_f32_bf16<<<(int)(NW / 4 + 255) / 256, 256, 0, stream>>>((const float4*)Wk, (uint2*)wkb, (int)(NW / 4));
    cvt_f32_bf16<<<(int)(NW / 4 + 255) / 256, 256, 0, stream>>>((const float4*)Wv, (uint2*)wvb, (int)(NW / 4));
    cvt_f32_bf16<<<(int)(NW / 4 + 255) / 256, 256, 0, stream>>>((const float4*)Wo, (uint2*)wob, (int)(NW / 4));

    dim3 gg(M / 64, C / 64), bb(128);
    gemm_bias<1><<<gg, bb, 0, stream>>>(xb, wqb, bq, qb,  M, C, C);
    gemm_bias<1><<<gg, bb, 0, stream>>>(xb, wkb, bk, kbf, M, C, C);
    gemm_bias<1><<<gg, bb, 0, stream>>>(xb, wvb, bv, vbf, M, C, C);
    flash_attn<<<dim3(T / 64, B * H), bb, 0, stream>>>(qb, kbf, vbf, yb, T, C);
    gemm_bias<0><<<gg, bb, 0, stream>>>(yb, wob, bo, (float*)d_out, M, C, C);
}
